// TripletRankingLoss_59906203844807
// MI455X (gfx1250) — compile-verified, matrix operand order checked by reading, not run on hardware
//
#include <hip/hip_runtime.h>
#include <hip/hip_bf16.h>

// Triplet ranking loss, B=8192, MARGIN=0.5, THRESH=0.1.
// Phase 1: async-stage targets[8192] (32KB) to LDS (ASYNCcnt path, CDNA5).
// Phase 2: per-row pos/neg counts via v_wmma_f32_16x16x32_f16 (indicator x ones).
// Phase 3: wave32 VALU scan for hash-based uniform argmax sampling per row.
// Phase 4: global float atomics -> finalize kernel.

#define NB 8192
#define NMARGIN 0.5f
#define NTHRESH 0.1f

typedef __attribute__((ext_vector_type(16))) _Float16 v16h;
typedef __attribute__((ext_vector_type(8)))  float    v8f;

__device__ __forceinline__ unsigned mix2(unsigned a, unsigned b, unsigned salt) {
    unsigned x = a * 0x9E3779B9u ^ b * 0x85EBCA6Bu ^ salt;
    x ^= x >> 16; x *= 0x7FEB352Du;
    x ^= x >> 15; x *= 0x846CA68Bu;
    x ^= x >> 16;
    return x;
}

__global__ void triplet_init(float* ws) {
    ws[0] = 0.0f;   // loss sum
    ws[1] = 0.0f;   // triplet count
}

__global__ __launch_bounds__(128) void triplet_main(const float* __restrict__ preds,
                                                    const float* __restrict__ targets,
                                                    float* __restrict__ ws) {
    __shared__ float sT[NB];      // full targets array, 32 KB
    __shared__ float sCp[128];    // per-anchor pos counts (this block)
    __shared__ float sCn[128];    // per-anchor neg counts (this block)

    const int tid  = (int)threadIdx.x;
    const int lane = tid & 31;
    const int wave = tid >> 5;
    const int i    = (int)blockIdx.x * 128 + tid;   // anchor index, < 8192

    // ---- Phase 1: async copy targets -> LDS (128 lanes x 16B x 16 issues = 32KB)
    {
        unsigned ldsBase = (unsigned)(size_t)(&sT[0]);
        unsigned off     = (unsigned)(tid * 16);
#pragma unroll
        for (int k = 0; k < 16; ++k) {
            unsigned lofs = ldsBase + off + (unsigned)(k * 2048);
            unsigned gofs = off + (unsigned)(k * 2048);
            asm volatile("global_load_async_to_lds_b128 %0, %1, %2"
                         :: "v"(lofs), "v"(gofs), "s"(targets)
                         : "memory");
        }
        asm volatile("s_wait_asynccnt 0" ::: "memory");
    }
    __syncthreads();

    const float ti = sT[i];

    // ---- Phase 2: pos/neg counts via WMMA.
    // Each wave owns anchors [blk*128 + wave*32, +32) = two 16-anchor M-tiles.
    // A (16x32 f16) = indicator matrix, B = ones => D[m][n] = count (replicated over n).
    {
        v16h ones;
#pragma unroll
        for (int e = 0; e < 16; ++e) ones[e] = (_Float16)1.0f;

        const int kb = (lane >> 4) * 8;   // K base for this lane half (A layout)
        const int m  = lane & 15;         // row within tile

#pragma unroll
        for (int tile = 0; tile < 2; ++tile) {
            const int tb = (int)blockIdx.x * 128 + wave * 32 + tile * 16;
            const float ta = sT[tb + m];
            v8f cp = {};
            v8f cn = {};
            for (int j0 = 0; j0 < NB; j0 += 32) {
                v16h ap, an;
#pragma unroll
                for (int e = 0; e < 16; ++e) {
                    // v16h element e -> K = (e>=8 ? 16 : 0) + kb + (e&7)  (ISA A layout)
                    int K = ((e >> 3) << 4) + kb + (e & 7);
                    float d = ta - sT[j0 + K];
                    ap[e] = (_Float16)((d < -NTHRESH) ? 1.0f : 0.0f);
                    an[e] = (_Float16)((d >  NTHRESH) ? 1.0f : 0.0f);
                }
                cp = __builtin_amdgcn_wmma_f32_16x16x32_f16(
                        false, ap, false, ones, (short)0, cp, false, false);
                cn = __builtin_amdgcn_wmma_f32_16x16x32_f16(
                        false, an, false, ones, (short)0, cn, false, false);
            }
            // Column n=0 of each lane-half carries the row counts: lanes 0 and 16
            // hold rows m = r + 8*(lane>>4) in accumulator element r.
            if ((lane & 15) == 0) {
                int lb = wave * 32 + tile * 16 + (lane >> 4) * 8;
#pragma unroll
                for (int r = 0; r < 8; ++r) {
                    sCp[lb + r] = cp[r];
                    sCn[lb + r] = cn[r];
                }
            }
        }
    }
    __syncthreads();

    // ---- Phase 3: uniform-random argmax sampling per row (hash-based).
    unsigned bp = 0u, bn = 0u;
    int pj = -1, nj = -1;
    const float4* sT4 = (const float4*)sT;
    for (int q = 0; q < NB / 4; ++q) {
        float4 t4 = sT4[q];
        float tj[4] = {t4.x, t4.y, t4.z, t4.w};
        int jb = q * 4;
#pragma unroll
        for (int s = 0; s < 4; ++s) {
            float d = ti - tj[s];
            int j = jb + s;
            if (d < -NTHRESH) {
                unsigned u = mix2((unsigned)i, (unsigned)j, 0x1234567u);
                if (pj < 0 || u > bp) { bp = u; pj = j; }
            }
            if (d > NTHRESH) {
                unsigned u = mix2((unsigned)i, (unsigned)j, 0x89ABCDEu);
                if (nj < 0 || u > bn) { bn = u; nj = j; }
            }
        }
    }

    // ---- Phase 4: per-anchor hinge + global accumulation.
    float cp = sCp[tid];
    float cn = sCn[tid];
    bool valid = (cp > 0.5f) && (cn > 0.5f) && (pj >= 0) && (nj >= 0);
    if (valid) {
        float per = fmaxf(NMARGIN - (preds[pj] - preds[nj]), 0.0f);
        atomicAdd(&ws[0], per);
        atomicAdd(&ws[1], 1.0f);
    }
}

__global__ void triplet_fin(const float* __restrict__ ws, float* __restrict__ out) {
    float s = ws[0];
    float n = ws[1];
    out[0] = (n > 0.0f) ? (s / fmaxf(n, 1.0f)) : 0.0f;
}

extern "C" void kernel_launch(void* const* d_in, const int* in_sizes, int n_in,
                              void* d_out, int out_size, void* d_ws, size_t ws_size,
                              hipStream_t stream) {
    const float* preds   = (const float*)d_in[0];
    const float* targets = (const float*)d_in[1];
    float* ws  = (float*)d_ws;
    float* out = (float*)d_out;

    triplet_init<<<1, 1, 0, stream>>>(ws);
    triplet_main<<<NB / 128, 128, 0, stream>>>(preds, targets, ws);
    triplet_fin<<<1, 1, 0, stream>>>(ws, out);
}
